// MultiHeadAttention_8839042695452
// MI455X (gfx1250) — compile-verified, matrix operand order checked by reading, not run on hardware
//
#include <hip/hip_runtime.h>

// ---------------------------------------------------------------------------
// Types
// ---------------------------------------------------------------------------
typedef __bf16 bf16_t;
typedef __attribute__((ext_vector_type(4)))  __bf16 v4bf;
typedef __attribute__((ext_vector_type(8)))  __bf16 v8bf;
typedef __attribute__((ext_vector_type(16))) __bf16 v16bf;
typedef __attribute__((ext_vector_type(4)))  float  v4f;
typedef __attribute__((ext_vector_type(8)))  float  v8f;

// int4 pointer types matching the async-LDS builtin signature
// (param types print as "__vector_size__(4*sizeof(int)) int __device__ *")
typedef int v4i_t __attribute__((vector_size(16)));
typedef __attribute__((address_space(1))) v4i_t as1_v4i;
typedef __attribute__((address_space(3))) v4i_t as3_v4i;

// Problem constants (match reference)
#define BATCH   2
#define SEQ     2048
#define DMODEL  1024
#define NHEAD   16
#define DK      64
#define MTOT    (BATCH * SEQ)        // 4096 rows for the big GEMMs

// ---------------------------------------------------------------------------
// CDNA5 async global->LDS copy (16B per lane), with safe fallback
// ---------------------------------------------------------------------------
#if defined(__has_builtin)
#  if __has_builtin(__builtin_amdgcn_global_load_async_to_lds_b128)
#    define USE_ASYNC_LDS 1
#  endif
#endif
#ifndef USE_ASYNC_LDS
#  define USE_ASYNC_LDS 0
#endif

static __device__ __forceinline__ void copy_g2l(const bf16_t* g, bf16_t* l) {
#if USE_ASYNC_LDS
    __builtin_amdgcn_global_load_async_to_lds_b128(
        (as1_v4i*)g, (as3_v4i*)l, 0, 0);
#else
    *(v8bf*)l = *(const v8bf*)g;
#endif
}

static __device__ __forceinline__ void async_fence() {
#if USE_ASYNC_LDS
#  if __has_builtin(__builtin_amdgcn_s_wait_asynccnt)
    __builtin_amdgcn_s_wait_asynccnt(0);
#  else
    asm volatile("s_wait_asynccnt 0" ::: "memory");
#  endif
#endif
}

// ---------------------------------------------------------------------------
// WMMA helpers
// ---------------------------------------------------------------------------
static __device__ __forceinline__ v8f wmma_bf16(v16bf a, v16bf b, v8f c) {
    // D = A(16x32 bf16) x B(32x16 bf16) + C(16x16 f32)
    return __builtin_amdgcn_wmma_f32_16x16x32_bf16(
        /*neg_a=*/false, a, /*neg_b=*/false, b,
        /*c_mod=*/(short)0, c, /*reuse_a=*/false, /*reuse_b=*/false);
}

// Load a 16x32 bf16 A/B fragment from an LDS tile (row-major, 'ld' elements
// per row, ld % 8 == 0 so every access is 16B aligned).
// CDNA5 16-bit A-matrix layout: lane (m = lane&15, half = lane>>4) holds
// K = {8h..8h+7} in VGPR0-3 and K = {16+8h..16+8h+7} in VGPR4-7:
// two contiguous 16B chunks -> two ds_load_b128.
static __device__ __forceinline__ v16bf load_frag(const bf16_t* base, int ld) {
    const int lane = threadIdx.x & 31;
    const int r    = lane & 15;
    const int kq   = (lane >> 4) << 3;   // 0 or 8
    const bf16_t* p = base + r * ld + kq;
    v8bf lo = *(const v8bf*)(p);
    v8bf hi = *(const v8bf*)(p + 16);
    return __builtin_shufflevector(lo, hi,
        0,1,2,3,4,5,6,7,8,9,10,11,12,13,14,15);
}

// ---------------------------------------------------------------------------
// Kernel 1: fp32 -> bf16 elementwise convert (4 elements / thread)
// ---------------------------------------------------------------------------
__global__ __launch_bounds__(256)
void cvt_f32_to_bf16(const float* __restrict__ src, bf16_t* __restrict__ dst, int n) {
    int i = (blockIdx.x * 256 + threadIdx.x) * 4;
    if (i + 3 < n) {
        v4f v = *(const v4f*)(src + i);
        v4bf o;
        o[0] = (__bf16)v[0];
        o[1] = (__bf16)v[1];
        o[2] = (__bf16)v[2];
        o[3] = (__bf16)v[3];
        *(v4bf*)(dst + i) = o;
    }
}

// ---------------------------------------------------------------------------
// Kernel 2: transpose + convert weight:  WT[n][k] = (bf16) W[k][n]
// ---------------------------------------------------------------------------
__global__ __launch_bounds__(256)
void transpose_cvt(const float* __restrict__ W, bf16_t* __restrict__ WT) {
    __shared__ float tile[32][33];
    const int tx = threadIdx.x;
    const int ty = threadIdx.y;
    const int x0 = blockIdx.x * 32;
    const int y0 = blockIdx.y * 32;
#pragma unroll
    for (int j = 0; j < 32; j += 8)
        tile[ty + j][tx] = W[(size_t)(y0 + ty + j) * DMODEL + (x0 + tx)];
    __syncthreads();
#pragma unroll
    for (int j = 0; j < 32; j += 8)
        WT[(size_t)(x0 + ty + j) * DMODEL + (y0 + tx)] = (bf16_t)tile[tx][ty + j];
}

// ---------------------------------------------------------------------------
// Kernel 3/5: bf16 WMMA GEMM  out[m][n] = sum_k A[m][k] * WT[n][k] + bias[n]
//   block tile 128(M) x 128(N), K-step 32, 8 waves (4x2), wave tile 32x64,
//   double-buffered LDS fed by async global->LDS DMA, prefetch of tile k+2.
// mode 0: bf16 out, layout [b,h,s,d]      (Q/K projections)
// mode 1: bf16 out, layout [b,h,d,s]      (V projection, pre-transposed)
// mode 2: f32  out, row-major [m][n]      (final output projection)
// ---------------------------------------------------------------------------
#define BM  128
#define BN  128
#define BK  32
#define LDT 48   // padded LDS stride (elements); 48*2=96B, 16B-aligned rows

__global__ __launch_bounds__(256)
void gemm_bf16(const bf16_t* __restrict__ A, const bf16_t* __restrict__ BT,
               const float* __restrict__ bias, void* __restrict__ out,
               int M, int N, int K, int mode)
{
    __shared__ bf16_t As[2][BM * LDT];
    __shared__ bf16_t Bs[2][BN * LDT];

    const int t    = threadIdx.x;
    const int w    = t >> 5;
    const int lane = t & 31;
    const int wm   = w >> 1;       // 0..3 -> rows 32*wm
    const int wn   = w & 1;        // 0..1 -> cols 64*wn
    const int m0   = blockIdx.x * BM;
    const int n0   = blockIdx.y * BN;

    v8f c[2][4] = {};

    const int NK = K / BK;

    auto load_tiles = [&](int kt, int buf) {
        // A tile: 128x32 bf16 = 512 16B chunks; B tile: 128x32 = 512 chunks
#pragma unroll
        for (int q = 0; q < 2; ++q) {
            int idx = t + q * 256;
            int row = idx >> 2;
            int col = (idx & 3) << 3;
            copy_g2l(A  + (size_t)(m0 + row) * K + kt * BK + col,
                     &As[buf][row * LDT + col]);
            copy_g2l(BT + (size_t)(n0 + row) * K + kt * BK + col,
                     &Bs[buf][row * LDT + col]);
        }
    };

    load_tiles(0, 0);
    async_fence();
    __syncthreads();

#pragma unroll 1
    for (int kt = 0; kt < NK; ++kt) {
        const int buf = kt & 1;
        if (kt + 1 < NK) load_tiles(kt + 1, buf ^ 1);
        if (kt + 2 < NK) {
            __builtin_prefetch((const void*)(A  + (size_t)(m0 + (t >> 1)) * K + (kt + 2) * BK), 0, 0);
            __builtin_prefetch((const void*)(BT + (size_t)(n0 + (t >> 1)) * K + (kt + 2) * BK), 0, 0);
        }

        const bf16_t* as = &As[buf][0];
        const bf16_t* bs = &Bs[buf][0];
        v16bf a0 = load_frag(as + (32 * wm)      * LDT, LDT);
        v16bf a1 = load_frag(as + (32 * wm + 16) * LDT, LDT);
        v16bf bf[4];
#pragma unroll
        for (int j = 0; j < 4; ++j)
            bf[j] = load_frag(bs + (64 * wn + 16 * j) * LDT, LDT);

#pragma unroll
        for (int j = 0; j < 4; ++j) {
            c[0][j] = wmma_bf16(a0, bf[j], c[0][j]);
            c[1][j] = wmma_bf16(a1, bf[j], c[1][j]);
        }

        async_fence();     // my DMA into buf^1 complete before group barrier
        __syncthreads();
    }

    // Epilogue. C/D layout: VGPR r -> row = r + 8*(lane>=16), col = lane&15.
    const int hh = lane >> 4;
    const int nl = lane & 15;
#pragma unroll
    for (int i = 0; i < 2; ++i)
#pragma unroll
        for (int j = 0; j < 4; ++j)
#pragma unroll
            for (int r = 0; r < 8; ++r) {
                int mg = m0 + 32 * wm + 16 * i + r + 8 * hh;
                int ng = n0 + 64 * wn + 16 * j + nl;
                float val = c[i][j][r] + bias[ng];
                if (mode == 2) {
                    ((float*)out)[(size_t)mg * N + ng] = val;
                } else {
                    int b = mg >> 11, s = mg & (SEQ - 1);
                    int h = ng >> 6,  d = ng & (DK - 1);
                    size_t off = (mode == 0)
                        ? ((((size_t)b * NHEAD + h) * SEQ + s) * DK + d)   // [b,h,s,d]
                        : ((((size_t)b * NHEAD + h) * DK + d) * SEQ + s);  // [b,h,d,s]
                    ((bf16_t*)out)[off] = (bf16_t)val;
                }
            }
}

// ---------------------------------------------------------------------------
// Kernel 4: flash attention over one (b,h), 128 query rows per workgroup.
//   qp : [b,h,s,d] bf16, kp : [b,h,s,d] bf16, vpt : [b,h,d,s] bf16
//   attn out: [b,s,h*64+d] bf16 (row-major 4096x1024 for the final GEMM)
// ---------------------------------------------------------------------------
#define QB   128
#define KBLK 64
#define LQK  72   // padded stride, 72*2=144B rows (16B-aligned)

__global__ __launch_bounds__(256)
void attention_fwd(const bf16_t* __restrict__ qp, const bf16_t* __restrict__ kp,
                   const bf16_t* __restrict__ vpt, bf16_t* __restrict__ attn)
{
    __shared__ bf16_t Qs[QB * LQK];          // 128 q-rows x 64 d
    __shared__ bf16_t Ks[KBLK * LQK];        // 64 keys  x 64 d
    __shared__ bf16_t Vt[KBLK * LQK];        // 64 d     x 64 keys
    __shared__ bf16_t Ps[8][16 * 64];        // per-wave P (16 q x 64 k)

    const int t     = threadIdx.x;
    const int w     = t >> 5;
    const int lane  = t & 31;
    const int qbase = blockIdx.x * QB;
    const int bh    = blockIdx.y;            // b*NHEAD + h
    const int b     = bh >> 4;
    const int h     = bh & (NHEAD - 1);

    const bf16_t* Qg = qp  + ((size_t)bh * SEQ + qbase) * DK;
    const bf16_t* Kg = kp  + (size_t)bh * SEQ * DK;
    const bf16_t* Vg = vpt + (size_t)bh * DK * SEQ;

    // Stage Q block via async DMA: 128x64 bf16 = 1024 chunks, 4 per thread
#pragma unroll
    for (int q = 0; q < 4; ++q) {
        int idx = t + q * 256;
        int row = idx >> 3;
        int col = (idx & 7) << 3;
        copy_g2l(Qg + (size_t)row * DK + col, &Qs[row * LQK + col]);
    }

    v8f o[4] = {};
    float mst[8], lst[8];
#pragma unroll
    for (int r = 0; r < 8; ++r) { mst[r] = -1e30f; lst[r] = 0.0f; }

    const int m0 = 16 * w;
    const int hh = lane >> 4;
    const int nl = lane & 15;

#pragma unroll 1
    for (int kt = 0; kt < SEQ / KBLK; ++kt) {
        __syncthreads();   // previous readers of Ks/Vt are done
        // Stage K (64x64, [s][d]) and Vt (64x64, [d][s]): 512 chunks each
#pragma unroll
        for (int q = 0; q < 2; ++q) {
            int idx = t + q * 256;
            int row = idx >> 3;
            int col = (idx & 7) << 3;
            copy_g2l(Kg + ((size_t)(kt * KBLK + row)) * DK + col,
                     &Ks[row * LQK + col]);
            copy_g2l(Vg + (size_t)row * SEQ + kt * KBLK + col,
                     &Vt[row * LQK + col]);
        }
        async_fence();     // covers Q staging on iter 0 as well
        __syncthreads();

        // scores S = Q_blk @ K_blk^T   (16 q-rows x 64 keys per wave)
        v8f c4[4] = {};
#pragma unroll
        for (int kd = 0; kd < DK; kd += 32) {
            v16bf aq = load_frag(&Qs[m0 * LQK + kd], LQK);
#pragma unroll
            for (int j = 0; j < 4; ++j) {
                v16bf bk = load_frag(&Ks[(16 * j) * LQK + kd], LQK);
                c4[j] = wmma_bf16(aq, bk, c4[j]);
            }
        }
#pragma unroll
        for (int j = 0; j < 4; ++j) c4[j] *= 0.125f;   // 1/sqrt(64)

        // online softmax: rows live in one 16-lane half, reduce via shfl_xor
#pragma unroll
        for (int r = 0; r < 8; ++r) {
            float rm = fmaxf(fmaxf(c4[0][r], c4[1][r]),
                             fmaxf(c4[2][r], c4[3][r]));
#pragma unroll
            for (int off = 8; off >= 1; off >>= 1)
                rm = fmaxf(rm, __shfl_xor(rm, off, 32));
            float mnew  = fmaxf(mst[r], rm);
            float alpha = __expf(mst[r] - mnew);
            mst[r] = mnew;
            float rs = 0.0f;
#pragma unroll
            for (int j = 0; j < 4; ++j) {
                float p = __expf(c4[j][r] - mnew);
                c4[j][r] = p;
                rs += p;
            }
#pragma unroll
            for (int off = 8; off >= 1; off >>= 1)
                rs += __shfl_xor(rs, off, 32);
            lst[r] = lst[r] * alpha + rs;
#pragma unroll
            for (int j = 0; j < 4; ++j) o[j][r] *= alpha;
        }

        // C-layout -> A-fragment layout via per-wave LDS round trip
        bf16_t* pw = &Ps[w][0];
#pragma unroll
        for (int j = 0; j < 4; ++j)
#pragma unroll
            for (int r = 0; r < 8; ++r)
                pw[(r + 8 * hh) * 64 + nl + 16 * j] = (bf16_t)c4[j][r];

        // O += P @ V  (B fragment n=d, k=s from Vt)
#pragma unroll
        for (int sd = 0; sd < KBLK; sd += 32) {
            v16bf ap = load_frag(pw + sd, 64);
#pragma unroll
            for (int j = 0; j < 4; ++j) {
                v16bf bv = load_frag(&Vt[(16 * j) * LQK + sd], LQK);
                o[j] = wmma_bf16(ap, bv, o[j]);
            }
        }
    }

    // normalize and store to [b, s, h*64 + d]
#pragma unroll
    for (int r = 0; r < 8; ++r) {
        float inv = 1.0f / lst[r];
        int sg = qbase + m0 + r + 8 * hh;
#pragma unroll
        for (int j = 0; j < 4; ++j) {
            float val = o[j][r] * inv;
            attn[((size_t)b * SEQ + sg) * DMODEL + h * DK + nl + 16 * j] =
                (bf16_t)val;
        }
    }
}

// ---------------------------------------------------------------------------
// Host-side orchestration
// ---------------------------------------------------------------------------
extern "C" void kernel_launch(void* const* d_in, const int* in_sizes, int n_in,
                              void* d_out, int out_size, void* d_ws, size_t ws_size,
                              hipStream_t stream)
{
    (void)in_sizes; (void)n_in; (void)out_size; (void)ws_size;

    const float* Q  = (const float*)d_in[0];
    const float* K  = (const float*)d_in[1];
    const float* V  = (const float*)d_in[2];
    const float* Wq = (const float*)d_in[3];
    const float* bq = (const float*)d_in[4];
    const float* Wk = (const float*)d_in[5];
    const float* bk = (const float*)d_in[6];
    const float* Wv = (const float*)d_in[7];
    const float* bv = (const float*)d_in[8];
    const float* Wo = (const float*)d_in[9];
    const float* bo = (const float*)d_in[10];

    char* ws = (char*)d_ws;
    size_t off = 0;
    auto alloc = [&](size_t bytes) -> char* {
        char* p = ws + off;
        off += (bytes + 255) & ~(size_t)255;
        return p;
    };

    const size_t XE = (size_t)MTOT * DMODEL;    // 4096*1024 activation elems
    const size_t WE = (size_t)DMODEL * DMODEL;  // weight elems

    bf16_t* Xq  = (bf16_t*)alloc(XE * 2);
    bf16_t* Xk  = (bf16_t*)alloc(XE * 2);
    bf16_t* Xv  = (bf16_t*)alloc(XE * 2);
    bf16_t* WqT = (bf16_t*)alloc(WE * 2);
    bf16_t* WkT = (bf16_t*)alloc(WE * 2);
    bf16_t* WvT = (bf16_t*)alloc(WE * 2);
    bf16_t* WoT = (bf16_t*)alloc(WE * 2);
    bf16_t* qp  = (bf16_t*)alloc(XE * 2);       // [b,h,s,d]
    bf16_t* kp  = (bf16_t*)alloc(XE * 2);       // [b,h,s,d]
    bf16_t* vpt = Xk;   // reuse: Xk dead after K projection
    bf16_t* attn = Xq;  // reuse: Xq dead after Q projection

    // 1) convert activations
    dim3 cb(256), cg((unsigned)(XE / (256 * 4)));
    cvt_f32_to_bf16<<<cg, cb, 0, stream>>>(Q, Xq, (int)XE);
    cvt_f32_to_bf16<<<cg, cb, 0, stream>>>(K, Xk, (int)XE);
    cvt_f32_to_bf16<<<cg, cb, 0, stream>>>(V, Xv, (int)XE);

    // 2) transpose-convert weights
    dim3 tb(32, 8), tg(DMODEL / 32, DMODEL / 32);
    transpose_cvt<<<tg, tb, 0, stream>>>(Wq, WqT);
    transpose_cvt<<<tg, tb, 0, stream>>>(Wk, WkT);
    transpose_cvt<<<tg, tb, 0, stream>>>(Wv, WvT);
    transpose_cvt<<<tg, tb, 0, stream>>>(Wo, WoT);

    // 3) QKV projections (V stored transposed per head)
    dim3 gb(256), gg(MTOT / BM, DMODEL / BN);
    gemm_bf16<<<gg, gb, 0, stream>>>(Xq, WqT, bq, qp,  MTOT, DMODEL, DMODEL, 0);
    gemm_bf16<<<gg, gb, 0, stream>>>(Xk, WkT, bk, kp,  MTOT, DMODEL, DMODEL, 0);
    gemm_bf16<<<gg, gb, 0, stream>>>(Xv, WvT, bv, vpt, MTOT, DMODEL, DMODEL, 1);

    // 4) attention
    dim3 ag(SEQ / QB, BATCH * NHEAD);
    attention_fwd<<<ag, dim3(256), 0, stream>>>(qp, kp, vpt, attn);

    // 5) output projection -> fp32 d_out
    gemm_bf16<<<gg, gb, 0, stream>>>(attn, WoT, bo, d_out, MTOT, DMODEL, DMODEL, 2);
}
